// DotAttention_88356067214138
// MI455X (gfx1250) — compile-verified, matrix operand order checked by reading, not run on hardware
//
#include <hip/hip_runtime.h>

typedef __attribute__((ext_vector_type(16))) _Float16 v16h;
typedef __attribute__((ext_vector_type(2)))  float    v2f;
typedef __attribute__((ext_vector_type(8)))  float    v8f;
typedef __attribute__((ext_vector_type(4)))  unsigned int v4u;
typedef __attribute__((ext_vector_type(8)))  int      v8i;
typedef __attribute__((ext_vector_type(4)))  int      v4i;

#define SEQ    2048
#define BATCH  32
#define HID    1024
#define SPLITS 32
#define ROWS   (SEQ / SPLITS)      // 64 rows per block
#define CHUNK  8                   // rows per TDM tile (f32 in LDS)
#define NCHUNK (ROWS / CHUNK)      // 8 chunks
#define TROW   1028                // dword row stride after TDM pad (1 dword per 256)

// ws layout (floats):
//   [0, ACC_ELEMS)                      : per-(b,split) accumulators [B][SPLITS][HID]
//   [ACC_ELEMS, ACC_ELEMS + B*SPLITS*2) : per-(b,split) (m, z)
#define ACC_ELEMS (BATCH * SPLITS * HID)

// TDM pads 1 dword after every 256 stored dwords -> element (r,h) lands here:
__device__ __forceinline__ int tidx(int r, int h) { return r * TROW + h + (h >> 8); }

__global__ __launch_bounds__(256)
void attn_partial_kernel(const float* __restrict__ rnn_out,
                         const float* __restrict__ state,
                         float* __restrict__ ws)
{
    const int b     = blockIdx.y;
    const int split = blockIdx.x;
    const int t     = threadIdx.x;
    const int wave  = t >> 5;
    const int lane  = t & 31;

    __shared__ float TILE[CHUNK * TROW];   // 8 rows f32, TDM-padded (32.9 KB)
    __shared__ float mvecf[HID];           // merged query vector, f32
    __shared__ float scpart[8][CHUNK];     // per-wave partial scores
    __shared__ float scores[CHUNK];

    // merged[b, h] = h < 512 ? state[1,0,b,h] : state[1,1,b,h-512]
    {
        const int h4 = t * 4;
        const float4 v = (t < 128)
            ? *(const float4*)(state + (size_t)(64 + b) * 512 + h4)
            : *(const float4*)(state + (size_t)(96 + b) * 512 + (h4 - 512));
        *(float4*)&mvecf[h4] = v;
    }

    float m_run = -INFINITY;
    float z_run = 0.0f;
    float acc0 = 0.f, acc1 = 0.f, acc2 = 0.f, acc3 = 0.f;
    const int hq = t * 4;
    const size_t row0 = (size_t)split * ROWS;

    // LDS byte address of TILE for the TDM descriptor (flat LDS addr = low 32 bits)
    const unsigned lds_base = (unsigned)(uintptr_t)&TILE[0];

    for (int c = 0; c < NCHUNK; ++c) {
        __syncthreads();   // previous tile fully consumed

        // ---- TDM: DMA 8 rows (stride B*H) of f32 into LDS, with pad-per-256-dwords ----
        if (wave == 0) {
            const size_t gaddr =
                (size_t)(uintptr_t)(rnn_out + ((row0 + (size_t)c * CHUNK) * BATCH + b) * HID);
            // D# group0: count=1 | lds_addr | 57b global_addr | type=2
            const v4u g0 = { 1u,
                             lds_base,
                             (unsigned)(gaddr & 0xFFFFFFFFu),
                             (unsigned)(gaddr >> 32) | (2u << 30) };
            // D# group1:
            //  d0: wg_mask=0 | data_size=2(4B)<<16 | pad_enable<<20 | pad_interval=7(256dw)<<22 | pad_amount=0(1dw)<<25
            //  d1: atomic_barrier_addr=0 [15:0] | tensor_dim0.lo16 << 16
            //  d2: tensor_dim0.hi16 [15:0] | tensor_dim1.lo16 << 16
            //  d3: tensor_dim1.hi16 [15:0] | tile_dim0 << 16
            //  d4: tile_dim1 [15:0] | tile_dim2=0 << 16
            //  d5: tensor_dim0_stride.lo32
            //  d6: stride0.hi16 | stride1.lo16 (0)   d7: stride1.hi32 (0)
            const v8i g1 = { (int)((2u << 16) | (1u << 20) | (7u << 22)),
                             (int)((unsigned)HID << 16),
                             (int)((unsigned)SEQ << 16),
                             (int)((unsigned)HID << 16),
                             CHUNK,
                             BATCH * HID,
                             0, 0 };
            const v4i g2 = { 0, 0, 0, 0 };
            const v4i g3 = { 0, 0, 0, 0 };
            const v8i g4 = { 0, 0, 0, 0, 0, 0, 0, 0 };
            __builtin_amdgcn_tensor_load_to_lds(g0, g1, g2, g3, g4, 0);
            __builtin_amdgcn_s_wait_tensorcnt(0);
        }
        __syncthreads();   // tile visible to all waves

        // prefetch next chunk rows toward L2 (global_prefetch_b8)
        if (c + 1 < NCHUNK) {
            const int pr = t >> 5;              // 8 rows
            const int pc = (t & 31) * 32;       // 32 segments x 128B
            __builtin_prefetch(
                rnn_out + ((row0 + (size_t)(c + 1) * CHUNK + pr) * BATCH + b) * HID + pc, 0, 1);
        }

        // ---- scores[8] = TILE[8 x 1024] * mvec via WMMA (waves split K) ----
        v8f d;
        #pragma unroll
        for (int e = 0; e < 8; ++e) d[e] = 0.0f;

        const int M = lane & 15;
        const int r = M & 7;                   // rows 8..15 duplicate rows 0..7 (ignored)
        const int koff = (lane >> 4) * 2;

#if __has_builtin(__builtin_amdgcn_wmma_f32_16x16x4_f32)
        // full-precision f32 WMMA, K-step 4; wave w covers k0 = w*4 + i*32
        #pragma unroll 8
        for (int i = 0; i < 32; ++i) {
            const int k0 = wave * 4 + i * 32;
            v2f a, bb;
            a.x  = TILE[tidx(r, k0 + koff)];
            a.y  = TILE[tidx(r, k0 + koff + 1)];
            bb.x = mvecf[k0 + koff];
            bb.y = mvecf[k0 + koff + 1];
            d = __builtin_amdgcn_wmma_f32_16x16x4_f32(
                    false, a, false, bb, (short)0, d, false, false);
        }
#else
        // fallback: convert f32 tile to f16 in registers, K-step 32
        #pragma unroll
        for (int i = 0; i < 4; ++i) {
            const int k0 = i * 256 + wave * 32;
            v16h a, bb;
            const int aoff = (lane >= 16) ? 8 : 0;
            #pragma unroll
            for (int e = 0; e < 16; ++e) {
                const int ka = k0 + aoff + e + ((e >= 8) ? 8 : 0);
                a[e] = (_Float16)TILE[tidx(r, ka)];
            }
            const int boff = (lane >= 16) ? 16 : 0;
            #pragma unroll
            for (int e = 0; e < 16; ++e)
                bb[e] = (_Float16)mvecf[k0 + boff + e];
            d = __builtin_amdgcn_wmma_f32_16x16x32_f16(
                    false, a, false, bb, (short)0, d, false, false);
        }
#endif
        // lane 0 holds D rows M=0..7 (the valid rows) in its 8 VGPRs
        if (lane == 0) {
            #pragma unroll
            for (int j = 0; j < 8; ++j) scpart[wave][j] = d[j];
        }
        __syncthreads();
        if (t < CHUNK) {
            float s = 0.f;
            #pragma unroll
            for (int wv = 0; wv < 8; ++wv) s += scpart[wv][t];
            scores[t] = s;
        }
        __syncthreads();

        // ---- online softmax update (redundant per thread) ----
        float w8[CHUNK];
        float m_c = -INFINITY;
        #pragma unroll
        for (int rr = 0; rr < CHUNK; ++rr) m_c = fmaxf(m_c, scores[rr]);
        const float m_new = fmaxf(m_run, m_c);
        const float scale = __expf(m_run - m_new);
        float zsum = 0.f;
        #pragma unroll
        for (int rr = 0; rr < CHUNK; ++rr) {
            w8[rr] = __expf(scores[rr] - m_new);
            zsum += w8[rr];
        }
        z_run = z_run * scale + zsum;
        m_run = m_new;

        acc0 *= scale; acc1 *= scale; acc2 *= scale; acc3 *= scale;
        #pragma unroll
        for (int rr = 0; rr < CHUNK; ++rr) {
            const float wr = w8[rr];
            const int base = tidx(rr, hq);      // hq % 4 == 0 -> no pad crossing
            acc0 += wr * TILE[base + 0];
            acc1 += wr * TILE[base + 1];
            acc2 += wr * TILE[base + 2];
            acc3 += wr * TILE[base + 3];
        }
    }

    float* wsacc = ws + (size_t)(b * SPLITS + split) * HID;
    float4 o; o.x = acc0; o.y = acc1; o.z = acc2; o.w = acc3;
    *(float4*)(wsacc + hq) = o;
    if (t == 0) {
        float* mz = ws + ACC_ELEMS + (size_t)(b * SPLITS + split) * 2;
        mz[0] = m_run;
        mz[1] = z_run;
    }
}

__global__ __launch_bounds__(256)
void attn_finalize_kernel(const float* __restrict__ ws, float* __restrict__ out)
{
    const int b = blockIdx.x;
    const int t = threadIdx.x;
    const float* mz = ws + ACC_ELEMS + (size_t)b * SPLITS * 2;

    __shared__ float f[SPLITS];

    float M = -INFINITY;
    #pragma unroll
    for (int i = 0; i < SPLITS; ++i) M = fmaxf(M, mz[2 * i]);
    float Z = 0.f;
    #pragma unroll
    for (int i = 0; i < SPLITS; ++i) Z += mz[2 * i + 1] * __expf(mz[2 * i] - M);
    const float inv = 1.0f / Z;
    if (t < SPLITS) f[t] = __expf(mz[2 * t] - M) * inv;
    __syncthreads();

    const int h = t * 4;
    float a0 = 0.f, a1 = 0.f, a2 = 0.f, a3 = 0.f;
    #pragma unroll 4
    for (int i = 0; i < SPLITS; ++i) {
        const float4 v = *(const float4*)(ws + (size_t)(b * SPLITS + i) * HID + h);
        const float fi = f[i];
        a0 += fi * v.x; a1 += fi * v.y; a2 += fi * v.z; a3 += fi * v.w;
    }
    float4 o; o.x = a0; o.y = a1; o.z = a2; o.w = a3;
    *(float4*)(out + (size_t)b * HID + h) = o;
}

extern "C" void kernel_launch(void* const* d_in, const int* in_sizes, int n_in,
                              void* d_out, int out_size, void* d_ws, size_t ws_size,
                              hipStream_t stream)
{
    const float* rnn_out = (const float*)d_in[0];
    const float* state   = (const float*)d_in[1];
    float* out = (float*)d_out;
    float* ws  = (float*)d_ws;

    attn_partial_kernel<<<dim3(SPLITS, BATCH), 256, 0, stream>>>(rnn_out, state, ws);
    attn_finalize_kernel<<<BATCH, 256, 0, stream>>>(ws, out);

    (void)in_sizes; (void)n_in; (void)out_size; (void)ws_size;
}
